// Block_20581483283089
// MI455X (gfx1250) — matrix operand order repaired and checked by reading, not compiled, not run on hardware
//
#include <hip/hip_runtime.h>
#include <hip/hip_fp16.h>
#include <hip/hip_bf16.h>

// ---------------------------------------------------------------------------
// Types for WMMA fragments (CDNA5 wave32, V_WMMA_F32_16X16X32_F16)
// ---------------------------------------------------------------------------
typedef __attribute__((ext_vector_type(16))) _Float16 v16h;
typedef __attribute__((ext_vector_type(8)))  _Float16 v8h;
typedef __attribute__((ext_vector_type(8)))  float    v8f;

union F16x16 {
    v16h v;
    v8h  h8[2];
    _Float16 h[16];
};

#define EMBED   768
#define HIDDEN  3072
#define QKV_N   2304
#define SEQ     2048
#define BATCH   2
#define ROWS    (BATCH * SEQ)   // 4096
#define HEADS   12
#define HDIM    64

// ---------------------------------------------------------------------------
// Fragment loaders, matching the ISA-documented 16-bit layouts.
//
// A (16x32, MxK): lane L (0..15) holds row M=L; element e maps to
//   k = (e<8 ? 0 : 16) + (lane>>4)*8 + (e&7)   -> two contiguous 16B loads
//
// B (32x16, KxN): lane (n or n+16) holds column N=n, k as for A: (e<8 ? 0 : 16) + (lane>>4)*8 + (e&7)
//   -> two contiguous 16B loads
// For out = A * W^T,  B[k][n] = W[n][k]: lane n reads W row n contiguously.
// ---------------------------------------------------------------------------
__device__ __forceinline__ v16h load_a_frag(const _Float16* p, int ld) {
    int lane = threadIdx.x & 31;
    int row  = lane & 15;
    int hi   = lane >> 4;
    const _Float16* q = p + (size_t)row * ld + hi * 8;
    F16x16 f;
    f.h8[0] = *(const v8h*)(q);        // k = hi*8 + 0..7
    f.h8[1] = *(const v8h*)(q + 16);   // k = 16 + hi*8 + 0..7
    return f.v;
}

__device__ __forceinline__ v16h load_b_frag(const _Float16* p, int ld) {
    int lane = threadIdx.x & 31;
    int col  = lane & 15;
    int hi   = lane >> 4;
    const _Float16* q = p + (size_t)col * ld + hi * 8;
    F16x16 f;
    f.h8[0] = *(const v8h*)(q);        // k = hi*8 + 0..7
    f.h8[1] = *(const v8h*)(q + 16);   // k = 16 + hi*8 + 0..7
    return f.v;
}

__device__ __forceinline__ v8f wmma32(v16h a, v16h b, v8f c) {
    return __builtin_amdgcn_wmma_f32_16x16x32_f16(
        /*neg_a=*/false, a, /*neg_b=*/false, b,
        /*c_mod=*/(short)0, c, /*reuse_a=*/false, /*reuse_b=*/false);
}

// ---------------------------------------------------------------------------
// DPP16 butterfly reductions over 16-lane groups (no LDS, no counters).
// Steps: xor1 (quad_perm 1,0,3,2), xor2 (quad_perm 2,3,0,1),
//        row_half_mirror (0x141), row_mirror (0x140).
// Each step combines two disjoint already-reduced groups -> valid butterfly.
// ---------------------------------------------------------------------------
template<int CTRL>
__device__ __forceinline__ float dpp_swap(float v) {
    int i = __builtin_bit_cast(int, v);
    int j = __builtin_amdgcn_update_dpp(i, i, CTRL, 0xF, 0xF, true);
    return __builtin_bit_cast(float, j);
}

__device__ __forceinline__ float red16_max(float x) {
    x = fmaxf(x, dpp_swap<0xB1>(x));    // quad_perm(1,0,3,2)
    x = fmaxf(x, dpp_swap<0x4E>(x));    // quad_perm(2,3,0,1)
    x = fmaxf(x, dpp_swap<0x141>(x));   // row_half_mirror
    x = fmaxf(x, dpp_swap<0x140>(x));   // row_mirror
    return x;
}

__device__ __forceinline__ float red16_sum(float x) {
    x += dpp_swap<0xB1>(x);
    x += dpp_swap<0x4E>(x);
    x += dpp_swap<0x141>(x);
    x += dpp_swap<0x140>(x);
    return x;
}

// ---------------------------------------------------------------------------
// fp32 -> fp16 cast (weights)
// ---------------------------------------------------------------------------
__global__ void cast_f16_kernel(const float* __restrict__ in,
                                _Float16* __restrict__ out, int n) {
    int i = blockIdx.x * 256 + threadIdx.x;
    if (i < n) out[i] = (_Float16)in[i];
}

// ---------------------------------------------------------------------------
// LayerNorm over rows of 768, fp32 in -> fp16 out. One block per row.
// ---------------------------------------------------------------------------
__global__ __launch_bounds__(256)
void ln_f16_kernel(const float* __restrict__ x, const float* __restrict__ w,
                   const float* __restrict__ b, _Float16* __restrict__ out) {
    int row = blockIdx.x;
    int tid = threadIdx.x;
    const float* xr = x + (size_t)row * EMBED;

    float v[3];
    float s = 0.0f, ss = 0.0f;
#pragma unroll
    for (int i = 0; i < 3; ++i) {
        v[i] = xr[tid + 256 * i];
        s  += v[i];
        ss += v[i] * v[i];
    }
    __shared__ float rs[256], rq[256];
    rs[tid] = s; rq[tid] = ss;
    __syncthreads();
    for (int off = 128; off > 0; off >>= 1) {
        if (tid < off) { rs[tid] += rs[tid + off]; rq[tid] += rq[tid + off]; }
        __syncthreads();
    }
    float mean = rs[0] * (1.0f / EMBED);
    float var  = rq[0] * (1.0f / EMBED) - mean * mean;
    float rstd = rsqrtf(var + 1e-5f);
#pragma unroll
    for (int i = 0; i < 3; ++i) {
        int idx = tid + 256 * i;
        float y = (v[i] - mean) * rstd * w[idx] + b[idx];
        out[(size_t)row * EMBED + idx] = (_Float16)y;
    }
}

// ---------------------------------------------------------------------------
// V transpose: qkv16[(b*SEQ+t)*2304 + 1536 + h*64 + d]
//           -> Vt[((b*HEADS+h)*HDIM + d)*SEQ + t]
// Grid: (SEQ/64, BATCH*HEADS), block 256. 64x64 f16 tile through padded LDS so
// both global read (along d) and global write (along t) are coalesced.
// ---------------------------------------------------------------------------
__global__ __launch_bounds__(256)
void transpose_v_kernel(const _Float16* __restrict__ qkv,
                        _Float16* __restrict__ vt) {
    __shared__ _Float16 tile[64][64 + 8];
    int bh = blockIdx.y;
    int b  = bh / HEADS;
    int h  = bh % HEADS;
    int t0 = blockIdx.x * 64;

    int d  = threadIdx.x & 63;        // 0..63
    int r0 = threadIdx.x >> 6;        // 0..3
    const _Float16* src = qkv + ((size_t)b * SEQ) * QKV_N + 2 * EMBED + h * HDIM;
#pragma unroll
    for (int i = 0; i < 16; ++i) {
        int t = r0 + 4 * i;           // 0..63
        tile[t][d] = src[(size_t)(t0 + t) * QKV_N + d];
    }
    __syncthreads();
    int t  = threadIdx.x & 63;
    _Float16* dst = vt + ((size_t)bh * HDIM) * SEQ + t0;
#pragma unroll
    for (int i = 0; i < 16; ++i) {
        int dd = r0 + 4 * i;
        dst[(size_t)dd * SEQ + t] = tile[t][dd];
    }
}

// ---------------------------------------------------------------------------
// Generic GEMM:  C[M,N] = act( A[M,K](f16) * W[N,K](f16)^T + bias + resid )
// Block = 128 threads (4 waves). Each wave: 32x64 tile = 2x4 WMMA accumulators.
// Grid: (N/64, M/128). M,N,K all multiples of {128,64,32} for our shapes.
// ---------------------------------------------------------------------------
template<bool GELU>
__global__ __launch_bounds__(128)
void gemm_wmma_kernel(const _Float16* __restrict__ A,
                      const _Float16* __restrict__ W,
                      const float* __restrict__ bias,
                      const float* __restrict__ resid,
                      float* __restrict__ outf,
                      _Float16* __restrict__ outh,
                      int M, int N, int K) {
    int wave = threadIdx.x >> 5;
    int lane = threadIdx.x & 31;
    int n0 = blockIdx.x * 64;
    int m0 = blockIdx.y * 128 + wave * 32;

    v8f acc[2][4] = {};

    for (int k = 0; k < K; k += 32) {
        v16h a0 = load_a_frag(A + (size_t)m0 * K + k, K);
        v16h a1 = load_a_frag(A + (size_t)(m0 + 16) * K + k, K);
        v16h b0 = load_b_frag(W + (size_t)(n0 +  0) * K + k, K);
        v16h b1 = load_b_frag(W + (size_t)(n0 + 16) * K + k, K);
        v16h b2 = load_b_frag(W + (size_t)(n0 + 32) * K + k, K);
        v16h b3 = load_b_frag(W + (size_t)(n0 + 48) * K + k, K);
        acc[0][0] = wmma32(a0, b0, acc[0][0]);
        acc[0][1] = wmma32(a0, b1, acc[0][1]);
        acc[0][2] = wmma32(a0, b2, acc[0][2]);
        acc[0][3] = wmma32(a0, b3, acc[0][3]);
        acc[1][0] = wmma32(a1, b0, acc[1][0]);
        acc[1][1] = wmma32(a1, b1, acc[1][1]);
        acc[1][2] = wmma32(a1, b2, acc[1][2]);
        acc[1][3] = wmma32(a1, b3, acc[1][3]);
    }

    // Epilogue. C layout: VGPR r holds row (r + 8*hi), col = lane&15.
    int coln   = lane & 15;
    int rowoff = (lane >> 4) * 8;
#pragma unroll
    for (int j = 0; j < 4; ++j) {
        int n = n0 + 16 * j + coln;
        float bv = bias[n];
#pragma unroll
        for (int i = 0; i < 2; ++i) {
#pragma unroll
            for (int r = 0; r < 8; ++r) {
                int m = m0 + 16 * i + rowoff + r;
                float val = acc[i][j][r] + bv;
                if (resid) val += resid[(size_t)m * N + n];
                if (GELU)  val = 0.5f * val * (1.0f + erff(val * 0.70710678118654752f));
                if (outf)  outf[(size_t)m * N + n] = val;
                if (outh)  outh[(size_t)m * N + n] = (_Float16)val;
            }
        }
    }
}

// ---------------------------------------------------------------------------
// Flash attention. qkv16: [B*T, 2304] f16 (Q|K|V each 768 = 12 heads * 64),
// vt: [B*H, 64, SEQ] f16 (transposed V). attn = softmax((Q K^T) * 8), o = attn V
// (reference MULTIPLIES by sqrt(head_dim) = 8; folded into Q fragments, exact
// since 8 is a power of two).
// Grid: (SEQ/64, BATCH*HEADS); block 128 (4 waves), each wave owns 16 query
// rows, streams keys in tiles of 32 with online softmax (DPP16 butterfly
// reductions, no LDS traffic). P tile bounces through per-wave LDS
// (C-layout -> A-layout). V fragments are contiguous two-b128 loads from vt.
// ---------------------------------------------------------------------------
__global__ __launch_bounds__(128)
void flash_attn_kernel(const _Float16* __restrict__ qkv,
                       const _Float16* __restrict__ vt,
                       _Float16* __restrict__ o16) {
    const int ld = QKV_N;
    int wave = threadIdx.x >> 5;
    int lane = threadIdx.x & 31;
    int col  = lane & 15;
    int hi   = lane >> 4;

    int bh = blockIdx.y;
    int b  = bh / HEADS;
    int h  = bh % HEADS;
    size_t base = (size_t)b * SEQ * ld;
    const _Float16* Qb = qkv + base + h * HDIM;
    const _Float16* Kb = qkv + base + EMBED + h * HDIM;
    const _Float16* Vt = vt + (size_t)bh * HDIM * SEQ;   // [64][SEQ]

    int q0 = blockIdx.x * 64 + wave * 16;

    // Q fragments for d = 0..31 and 32..63, pre-scaled by 8 (= sqrt(HDIM))
    v16h aq0 = load_a_frag(Qb + (size_t)q0 * ld + 0,  ld) * (_Float16)8.0;
    v16h aq1 = load_a_frag(Qb + (size_t)q0 * ld + 32, ld) * (_Float16)8.0;

    v8f o[4] = {};
    float mrun[8], lrun[8];
#pragma unroll
    for (int r = 0; r < 8; ++r) { mrun[r] = -1e30f; lrun[r] = 0.0f; }

    __shared__ alignas(16) _Float16 pbuf[4][16][32];

    for (int j0 = 0; j0 < SEQ; j0 += 32) {
        // S tiles: 16 queries x 32 keys (two 16-key WMMA tiles, K-dim = d)
        v8f s0 = {}, s1 = {};
        s0 = wmma32(aq0, load_b_frag(Kb + (size_t)j0 * ld +  0, ld), s0);
        s0 = wmma32(aq1, load_b_frag(Kb + (size_t)j0 * ld + 32, ld), s0);
        s1 = wmma32(aq0, load_b_frag(Kb + (size_t)(j0 + 16) * ld +  0, ld), s1);
        s1 = wmma32(aq1, load_b_frag(Kb + (size_t)(j0 + 16) * ld + 32, ld), s1);

        // Online softmax. Row r+8*hi lives in this lane; its 32 columns are
        // spread over the 16 lanes of this half-group (two tiles).
#pragma unroll
        for (int r = 0; r < 8; ++r) {
            float a = s0[r];
            float c = s1[r];
            float mt = red16_max(fmaxf(a, c));
            float mn   = fmaxf(mrun[r], mt);
            float corr = __expf(mrun[r] - mn);
            mrun[r] = mn;
            float p0 = __expf(a - mn);
            float p1 = __expf(c - mn);
            float rs = red16_sum(p0 + p1);
            lrun[r] = lrun[r] * corr + rs;
#pragma unroll
            for (int t = 0; t < 4; ++t) o[t][r] *= corr;
            // C-layout -> LDS (row-major 16x32 P tile, per-wave private)
            pbuf[wave][r + 8 * hi][col]      = (_Float16)p0;
            pbuf[wave][r + 8 * hi][16 + col] = (_Float16)p1;
        }

        // P as A-fragment (16x32, ld=32); V tiles as B-fragments from Vt:
        // B[k=j][n=d] = Vt[d0+n][j0+k]  -> contiguous two-b128 loads per lane.
        v16h ap = load_a_frag(&pbuf[wave][0][0], 32);
#pragma unroll
        for (int t = 0; t < 4; ++t) {
            v16h bv = load_b_frag(Vt + (size_t)(16 * t) * SEQ + j0, SEQ);
            o[t] = wmma32(ap, bv, o[t]);
        }
    }

    // Normalize and store into [B*T, 768] at column h*64
#pragma unroll
    for (int r = 0; r < 8; ++r) {
        float inv = 1.0f / lrun[r];
        int row = q0 + r + 8 * hi;
        size_t out = ((size_t)b * SEQ + row) * EMBED + h * HDIM;
#pragma unroll
        for (int t = 0; t < 4; ++t)
            o16[out + 16 * t + col] = (_Float16)(o[t][r] * inv);
    }
}

// ---------------------------------------------------------------------------
// Host-side orchestration
// ---------------------------------------------------------------------------
extern "C" void kernel_launch(void* const* d_in, const int* in_sizes, int n_in,
                              void* d_out, int out_size, void* d_ws, size_t ws_size,
                              hipStream_t stream) {
    const float* x      = (const float*)d_in[0];
    const float* ln1_w  = (const float*)d_in[1];
    const float* ln1_b  = (const float*)d_in[2];
    const float* qkv_w  = (const float*)d_in[3];
    const float* qkv_b  = (const float*)d_in[4];
    const float* proj_w = (const float*)d_in[5];
    const float* proj_b = (const float*)d_in[6];
    const float* ln2_w  = (const float*)d_in[7];
    const float* ln2_b  = (const float*)d_in[8];
    const float* fc1_w  = (const float*)d_in[9];
    const float* fc1_b  = (const float*)d_in[10];
    const float* fc2_w  = (const float*)d_in[11];
    const float* fc2_b  = (const float*)d_in[12];
    float* out = (float*)d_out;

    char* ws = (char*)d_ws;
    size_t off = 0;
    auto alloc = [&](size_t bytes) -> void* {
        void* p = ws + off;
        off = (off + bytes + 255) & ~(size_t)255;
        return p;
    };

    _Float16* Wq16  = (_Float16*)alloc((size_t)QKV_N  * EMBED  * 2);
    _Float16* Wp16  = (_Float16*)alloc((size_t)EMBED  * EMBED  * 2);
    _Float16* W116  = (_Float16*)alloc((size_t)HIDDEN * EMBED  * 2);
    _Float16* W216  = (_Float16*)alloc((size_t)EMBED  * HIDDEN * 2);
    _Float16* h16   = (_Float16*)alloc((size_t)ROWS * EMBED  * 2);
    _Float16* qkv16 = (_Float16*)alloc((size_t)ROWS * QKV_N  * 2);
    _Float16* vtb   = (_Float16*)alloc((size_t)BATCH * HEADS * HDIM * SEQ * 2);
    _Float16* o16   = (_Float16*)alloc((size_t)ROWS * EMBED  * 2);
    float*    x1    = (float*)   alloc((size_t)ROWS * EMBED  * 4);
    _Float16* h216  = (_Float16*)alloc((size_t)ROWS * EMBED  * 2);
    _Float16* mlp16 = (_Float16*)alloc((size_t)ROWS * HIDDEN * 2);

    auto cast = [&](const float* src, _Float16* dst, int n) {
        cast_f16_kernel<<<(n + 255) / 256, 256, 0, stream>>>(src, dst, n);
    };
    cast(qkv_w,  Wq16, QKV_N  * EMBED);
    cast(proj_w, Wp16, EMBED  * EMBED);
    cast(fc1_w,  W116, HIDDEN * EMBED);
    cast(fc2_w,  W216, EMBED  * HIDDEN);

    // ---- attention branch ----
    ln_f16_kernel<<<ROWS, 256, 0, stream>>>(x, ln1_w, ln1_b, h16);

    gemm_wmma_kernel<false><<<dim3(QKV_N / 64, ROWS / 128), 128, 0, stream>>>(
        h16, Wq16, qkv_b, nullptr, nullptr, qkv16, ROWS, QKV_N, EMBED);

    transpose_v_kernel<<<dim3(SEQ / 64, BATCH * HEADS), 256, 0, stream>>>(qkv16, vtb);

    flash_attn_kernel<<<dim3(SEQ / 64, BATCH * HEADS), 128, 0, stream>>>(qkv16, vtb, o16);

    gemm_wmma_kernel<false><<<dim3(EMBED / 64, ROWS / 128), 128, 0, stream>>>(
        o16, Wp16, proj_b, x, x1, nullptr, ROWS, EMBED, EMBED);

    // ---- MLP branch ----
    ln_f16_kernel<<<ROWS, 256, 0, stream>>>(x1, ln2_w, ln2_b, h216);

    gemm_wmma_kernel<true><<<dim3(HIDDEN / 64, ROWS / 128), 128, 0, stream>>>(
        h216, W116, fc1_b, nullptr, nullptr, mlp16, ROWS, HIDDEN, EMBED);

    gemm_wmma_kernel<false><<<dim3(EMBED / 64, ROWS / 128), 128, 0, stream>>>(
        mlp16, W216, fc2_b, x1, out, nullptr, ROWS, EMBED, HIDDEN);
}
